// GCN_49108656063298
// MI455X (gfx1250) — matrix operand order repaired and checked by reading, not compiled, not run on hardware
//
#include <hip/hip_runtime.h>
#include <cstdint>

#define NUM_USER 50000
#define NUM_ITEM 50000
#define NN (NUM_USER + NUM_ITEM)
#define DLAT 128
#define DFEAT 4096
#define DHID 512

typedef __attribute__((ext_vector_type(16))) __bf16 v16bf;
typedef __attribute__((ext_vector_type(8))) float v8f;

union FragBF { v16bf v; uint4 q[2]; };

__device__ __forceinline__ uint16_t f2bf(float f) {
    uint32_t u = __float_as_uint(f);
    uint32_t r = u + 0x7FFFu + ((u >> 16) & 1u);   // round-to-nearest-even
    return (uint16_t)(r >> 16);
}
__device__ __forceinline__ uint32_t pk2bf(float a, float b) {
    return (uint32_t)f2bf(a) | ((uint32_t)f2bf(b) << 16);
}

// Async 16B global -> LDS copy (per-lane addresses), tracked by ASYNCcnt.
__device__ __forceinline__ void async_copy_b128(uint16_t* lds_dst, const uint16_t* gsrc) {
    uint32_t lds = (uint32_t)(uintptr_t)lds_dst;           // 32-bit LDS address of the destination
    asm volatile("global_load_async_to_lds_b128 %0, %1, off"
                 :: "v"(lds), "v"(gsrc) : "memory");
}
__device__ __forceinline__ void wait_async_zero() {
    asm volatile("s_wait_asynccnt 0x0" ::: "memory");
}

// ---------------------------------------------------------------------------
// Tiled bf16 WMMA GEMM:  C[M x N] = act(A[M x K] * B[K x N] + bias)
// A: row-major (f32 converted on the fly, or bf16). B pre-transposed bf16 [N][K].
// Block tile: 128 x NT, 256 threads = 8 waves (4 M-groups x 2 N-groups).
// Wave tile: 32 x (NT/2) -> 2 x (NT/32) accumulators of v_wmma_f32_16x16x32_bf16.
// B tile staged with GLOBAL_LOAD_ASYNC_TO_LDS_B128 (ASYNCcnt path).
// ---------------------------------------------------------------------------
template <bool A_F32, int NT, bool RELU, bool OUT_BF16>
__global__ __launch_bounds__(256) void gemm_wmma_bf16(
    const float* __restrict__ Af, const uint16_t* __restrict__ Ab,
    const uint16_t* __restrict__ Bt, const float* __restrict__ bias,
    uint16_t* __restrict__ outB, float* __restrict__ outF,
    int M, int K, int ldo)
{
    constexpr int LDK = 40;            // padded K stride in LDS (80B = 20 banks)
    constexpr int WCOLS = NT / 2;      // columns per wave
    constexpr int NFRAG = WCOLS / 16;  // 16x16 frags per wave in N

    __shared__ uint16_t Als[128 * LDK];
    __shared__ uint16_t Bls[NT * LDK];

    const int tid  = threadIdx.x;
    const int lane = tid & 31;
    const int wave = tid >> 5;
    const int wm   = wave & 3;         // 4 wave rows
    const int wn   = wave >> 2;        // 2 wave cols
    const int rowBase = blockIdx.x * 128;
    const int colBase = blockIdx.y * NT;

    const v8f vzero = {0.f, 0.f, 0.f, 0.f, 0.f, 0.f, 0.f, 0.f};
    v8f acc[2][NFRAG];
#pragma unroll
    for (int i = 0; i < 2; ++i)
#pragma unroll
        for (int j = 0; j < NFRAG; ++j) acc[i][j] = vzero;

    const int mA  = tid >> 1;          // 128 rows, 2 threads/row
    const int khA = (tid & 1) * 16;    // which 16-element K half

    for (int kk = 0; kk < K; kk += 32) {
        // ---- stage B tile (NT x 32) into LDS via async DMA copy ----
#pragma unroll
        for (int u = 0; u < NT / 64; ++u) {
            const int q  = tid * (NT / 64) + u;
            const int n  = q >> 2;
            const int kq = q & 3;
            async_copy_b128(&Bls[n * LDK + kq * 8],
                            Bt + (size_t)(colBase + n) * K + kk + kq * 8);
        }
        // ---- stage A tile (128 x 32) into LDS as bf16, K-contiguous ----
        {
            const int row = rowBase + mA;
            uint4 q0 = {0u, 0u, 0u, 0u}, q1 = {0u, 0u, 0u, 0u};
            if (row < M) {
                if (A_F32) {
                    const float4* p = (const float4*)(Af + (size_t)row * K + kk + khA);
                    float4 v0 = p[0], v1 = p[1], v2 = p[2], v3 = p[3];
                    q0.x = pk2bf(v0.x, v0.y); q0.y = pk2bf(v0.z, v0.w);
                    q0.z = pk2bf(v1.x, v1.y); q0.w = pk2bf(v1.z, v1.w);
                    q1.x = pk2bf(v2.x, v2.y); q1.y = pk2bf(v2.z, v2.w);
                    q1.z = pk2bf(v3.x, v3.y); q1.w = pk2bf(v3.z, v3.w);
                    if (kk + 32 < K)   // prefetch next K-slice of the fat A stream
                        __builtin_prefetch(Af + (size_t)row * K + kk + 32 + khA, 0, 0);
                } else {
                    const uint4* p = (const uint4*)(Ab + (size_t)row * K + kk + khA);
                    q0 = p[0]; q1 = p[1];
                }
            }
            *(uint4*)&Als[mA * LDK + khA]     = q0;
            *(uint4*)&Als[mA * LDK + khA + 8] = q1;
        }
        wait_async_zero();             // B tile DMA complete
        __syncthreads();

        // ---- fragment loads + WMMA ----
        const int koffA = (lane >> 4) << 3;   // A: hi lanes hold K 8..15 / 24..31
        const int koffB = (lane >> 4) << 3;   // B: hi lanes hold K 8..15 / 24..31
        FragBF a0, a1;
        {
            const int m0 = wm * 32 + (lane & 15);
            a0.q[0] = *(const uint4*)&Als[m0 * LDK + koffA];
            a0.q[1] = *(const uint4*)&Als[m0 * LDK + koffA + 16];
            const int m1 = m0 + 16;
            a1.q[0] = *(const uint4*)&Als[m1 * LDK + koffA];
            a1.q[1] = *(const uint4*)&Als[m1 * LDK + koffA + 16];
        }
#pragma unroll
        for (int j = 0; j < NFRAG; ++j) {
            FragBF b;
            const int n = wn * WCOLS + j * 16 + (lane & 15);
            b.q[0] = *(const uint4*)&Bls[n * LDK + koffB];
            b.q[1] = *(const uint4*)&Bls[n * LDK + koffB + 16];
            acc[0][j] = __builtin_amdgcn_wmma_f32_16x16x32_bf16(
                false, a0.v, false, b.v, (short)0, acc[0][j], false, false);
            acc[1][j] = __builtin_amdgcn_wmma_f32_16x16x32_bf16(
                false, a1.v, false, b.v, (short)0, acc[1][j], false, false);
        }
        __syncthreads();
    }

    // ---- epilogue: bias (+ leaky relu) and store ----
    const int rhi = (lane >> 4) << 3;  // C/D: hi lanes hold rows M+8..M+15
#pragma unroll
    for (int i = 0; i < 2; ++i) {
#pragma unroll
        for (int j = 0; j < NFRAG; ++j) {
            const int col = colBase + wn * WCOLS + j * 16 + (lane & 15);
            const float bv = bias[col];
#pragma unroll
            for (int v = 0; v < 8; ++v) {
                const int r = rowBase + wm * 32 + i * 16 + rhi + v;
                if (r < M) {
                    float x = acc[i][j][v] + bv;
                    if (RELU) x = x > 0.f ? x : 0.01f * x;
                    if (OUT_BF16) outB[(size_t)r * ldo + col] = f2bf(x);
                    else          outF[(size_t)r * ldo + col] = x;
                }
            }
        }
    }
}

// ---------------------------------------------------------------------------
// Helper kernels
// ---------------------------------------------------------------------------
__global__ void k_transpose_cvt(const float* __restrict__ src, uint16_t* __restrict__ dst,
                                int K, int N) {  // src [K][N] f32 -> dst [N][K] bf16
    int idx = blockIdx.x * blockDim.x + threadIdx.x;
    if (idx >= K * N) return;
    int k = idx / N, n = idx - k * N;
    dst[(size_t)n * K + k] = f2bf(src[idx]);
}

__global__ void k_zero(float* p, int n) {
    int i = blockIdx.x * blockDim.x + threadIdx.x;
    if (i < n) p[i] = 0.f;
}

__global__ void k_copy_pref(const float* __restrict__ pref, float* __restrict__ xbuf,
                            float* __restrict__ out2, int n) {
    int i = blockIdx.x * blockDim.x + threadIdx.x;
    if (i < n) { float v = pref[i]; xbuf[i] = v; out2[i] = v; }
}

// warp-per-row L2 normalize; also writes normalized x into out (out = x term)
__global__ void k_normalize(float* __restrict__ x, float* __restrict__ out, int nRows) {
    int warp = (blockIdx.x * blockDim.x + threadIdx.x) >> 5;
    int lane = threadIdx.x & 31;
    if (warp >= nRows) return;
    float4* rp = (float4*)(x + (size_t)warp * DLAT);
    float4 v = rp[lane];
    float ss = v.x * v.x + v.y * v.y + v.z * v.z + v.w * v.w;
    for (int o = 16; o > 0; o >>= 1) ss += __shfl_xor(ss, o);
    float s = 1.0f / fmaxf(sqrtf(ss), 1e-12f);
    v.x *= s; v.y *= s; v.z *= s; v.w *= s;
    rp[lane] = v;
    ((float4*)(out + (size_t)warp * DLAT))[lane] = v;
}

__global__ void k_degree(const int* __restrict__ rowP, float* __restrict__ deg, int E) {
    int e = blockIdx.x * blockDim.x + threadIdx.x;
    if (e < E) atomicAdd(&deg[rowP[e]], 1.0f);
}

__global__ void k_rsqrt(float* deg, int n) {
    int i = blockIdx.x * blockDim.x + threadIdx.x;
    if (i < n) deg[i] = rsqrtf(deg[i]);
}

// one edge x 4 floats per thread; norm computed on the fly from deg^{-1/2}
__global__ void k_scatter(const float* __restrict__ h, float* __restrict__ hn,
                          const int* __restrict__ rowP, const int* __restrict__ colP,
                          const float* __restrict__ dinv, int E) {
    int idx = blockIdx.x * blockDim.x + threadIdx.x;
    int e = idx >> 5;
    if (e >= E) return;
    int t = idx & 31;
    int r = rowP[e], c = colP[e];
    float wgt = dinv[r] * dinv[c];
    float4 v = ((const float4*)(h + (size_t)r * DLAT))[t];
    float* d = hn + (size_t)c * DLAT + t * 4;
    atomicAdd(d + 0, wgt * v.x);
    atomicAdd(d + 1, wgt * v.y);
    atomicAdd(d + 2, wgt * v.z);
    atomicAdd(d + 3, wgt * v.w);
}

__global__ void k_accum(float* __restrict__ out, const float* __restrict__ h, int n) {
    int i = blockIdx.x * blockDim.x + threadIdx.x;
    if (i < n) out[i] += h[i];
}

// ---------------------------------------------------------------------------
extern "C" void kernel_launch(void* const* d_in, const int* in_sizes, int n_in,
                              void* d_out, int out_size, void* d_ws, size_t ws_size,
                              hipStream_t stream) {
    const float* features   = (const float*)d_in[0];
    const float* preference = (const float*)d_in[1];
    const float* W1 = (const float*)d_in[2];
    const float* b1 = (const float*)d_in[3];
    const float* W2 = (const float*)d_in[4];
    const float* b2 = (const float*)d_in[5];
    const int*   ei = (const int*)d_in[6];
    const int E = in_sizes[6] / 2;
    const int* rowP = ei;
    const int* colP = ei + E;

    // workspace carve (all 16B aligned)
    char* w = (char*)d_ws;
    float*    xbuf   = (float*)w;                          // 100000*128 f32 = 51.2 MB
    float*    hbuf   = (float*)(w + 51200000);             // 51.2 MB (aliased as hidden bf16)
    uint16_t* hidden = (uint16_t*)hbuf;                    // 50000*512 bf16 = 51.2 MB
    uint16_t* w1t    = (uint16_t*)(w + 102400000);         // 512*4096 bf16 = 4 MB
    uint16_t* w2t    = (uint16_t*)(w + 106594304);         // 128*512 bf16
    float*    deg    = (float*)(w + 106725376);            // 100000 f32

    float* outMain = (float*)d_out;                        // [100000,128]
    float* outPref = outMain + (size_t)NN * DLAT;          // [50000,128] (tuple elem 2)

    const int T = 256;
    const int nNode = NN * DLAT;                           // 12.8M
    const int nPref = NUM_USER * DLAT;                     // 6.4M
    const int mBlocks = (NUM_USER + 127) / 128;            // 391

    // 1) weights -> bf16, transposed [N][K]
    k_transpose_cvt<<<(DFEAT * DHID + T - 1) / T, T, 0, stream>>>(W1, w1t, DFEAT, DHID);
    k_transpose_cvt<<<(DHID * DLAT + T - 1) / T, T, 0, stream>>>(W2, w2t, DHID, DLAT);

    // 2) hidden = leaky_relu(features @ W1 + b1)  [bf16 out]
    gemm_wmma_bf16<true, 256, true, true><<<dim3(mBlocks, DHID / 256), T, 0, stream>>>(
        features, nullptr, w1t, b1, hidden, nullptr, NUM_USER, DFEAT, DHID);

    // 3) item_feat = hidden @ W2 + b2  -> xbuf rows [NUM_USER, NN)
    gemm_wmma_bf16<false, 128, false, false><<<dim3(mBlocks, 1), T, 0, stream>>>(
        nullptr, hidden, w2t, b2, nullptr, xbuf + (size_t)NUM_USER * DLAT,
        NUM_USER, DHID, DLAT);

    // 4) x[0:NUM_USER] = preference; also emit tuple's preference output
    k_copy_pref<<<(nPref + T - 1) / T, T, 0, stream>>>(preference, xbuf, outPref, nPref);

    // 5) row-normalize x in place; out = x
    k_normalize<<<(NN * 32 + T - 1) / T, T, 0, stream>>>(xbuf, outMain, NN);

    // 6) degrees -> deg^{-1/2}
    k_zero<<<(NN + T - 1) / T, T, 0, stream>>>(deg, NN);
    k_degree<<<(E + T - 1) / T, T, 0, stream>>>(rowP, deg, E);
    k_rsqrt<<<(NN + T - 1) / T, T, 0, stream>>>(deg, NN);

    // 7) layer 1: h1 = scatter(x); out += h1
    k_zero<<<(nNode + T - 1) / T, T, 0, stream>>>(hbuf, nNode);
    {
        long long thr = (long long)E * 32;
        k_scatter<<<(int)((thr + T - 1) / T), T, 0, stream>>>(xbuf, hbuf, rowP, colP, deg, E);
    }
    k_accum<<<(nNode + T - 1) / T, T, 0, stream>>>(outMain, hbuf, nNode);

    // 8) layer 2: h2 = scatter(h1); out += h2   (xbuf reused as h2)
    k_zero<<<(nNode + T - 1) / T, T, 0, stream>>>(xbuf, nNode);
    {
        long long thr = (long long)E * 32;
        k_scatter<<<(int)((thr + T - 1) / T), T, 0, stream>>>(hbuf, xbuf, rowP, colP, deg, E);
    }
    k_accum<<<(nNode + T - 1) / T, T, 0, stream>>>(outMain, xbuf, nNode);

    (void)n_in; (void)out_size; (void)ws_size; (void)in_sizes;
}
